// TwoBlockStructuredScoreNet_57870389347043
// MI455X (gfx1250) — matrix operand order repaired and checked by reading, not compiled, not run on hardware
//
#include <hip/hip_runtime.h>
#include <hip/hip_bf16.h>

#define N_DIM 512
#define H_DIM 64
#define B_DIM 32768
#define WAVES 4
#define NTHREADS (WAVES * 32)

typedef __attribute__((ext_vector_type(16))) __bf16 v16bf;
typedef __attribute__((ext_vector_type(8)))  float  v8f;

union FragU {
    v16bf v;
    uint4 q[2];
    unsigned int u[8];
};

__device__ __forceinline__ unsigned short f2bf(float f) {
    union { float f; unsigned int u; } x;
    x.f = f;
    unsigned int u = x.u;
    u += 0x7FFFu + ((u >> 16) & 1u);   // round-to-nearest-even
    return (unsigned short)(u >> 16);
}

__device__ __forceinline__ v8f wmma_bf16(v16bf a, v16bf b, v8f c) {
    return __builtin_amdgcn_wmma_f32_16x16x32_bf16(
        false, a, false, b, (short)0, c, false, false);
}

// A-fragment (16x32 bf16, MxK) from a row-major bf16 LDS tile, as 2x b128.
// A layout: lane L row M=L%16; lanes 0-15: K0..15, lanes 16-31: K16..31
// (contiguous, matching lds_bfrag). All addresses 16B-aligned by construction.
__device__ __forceinline__ v16bf lds_afrag(const unsigned short* base, int stride,
                                           int lane, int k0) {
    const int r  = lane & 15;
    const int hi = (lane >> 4) & 1;
    const unsigned short* p = base + r * stride + k0 + hi * 16;
    FragU f;
    f.q[0] = *(const uint4*)p;          // K = +0..7
    f.q[1] = *(const uint4*)(p + 8);    // K = +8..15
    return f.v;
}

// B-fragment (32x16 bf16, KxN) from an LDS panel P with P[n][k] = B[k][n]
// (k contiguous). Lane L: col N = n0 + L%16; lanes 0-15: K=k0..k0+15,
// lanes 16-31: K=k0+16..k0+31. 32B contiguous -> 2x b128.
__device__ __forceinline__ v16bf lds_bfrag(const unsigned short* P, int Kdim,
                                           int lane, int n0, int k0) {
    const int n  = n0 + (lane & 15);
    const int hi = (lane >> 4) & 1;
    const uint4* p = (const uint4*)(P + n * Kdim + k0 + hi * 16);
    FragU f;
    f.q[0] = p[0];
    f.q[1] = p[1];
    return f.v;
}

// Synchronous cooperative global->LDS panel copy (b128 vectors).
__device__ __forceinline__ void stage_panel(unsigned short* dst,
                                            const unsigned short* __restrict__ src,
                                            int u16count, int tid) {
    const uint4* s = (const uint4*)src;
    uint4*       d = (uint4*)dst;
    const int    n = u16count >> 3;
    for (int i = tid; i < n; i += NTHREADS) d[i] = s[i];
}

// Async DMA global->LDS panel copy (gfx1250 GLOBAL_LOAD_ASYNC_TO_LDS_B128,
// tracked by ASYNCcnt). 16B per lane per issue.
__device__ __forceinline__ void stage_panel_async(unsigned short* dst,
                                                  const unsigned short* __restrict__ src,
                                                  int u16count, int tid) {
    const unsigned int lds_base = (unsigned int)(unsigned long long)dst;
    const char* g = (const char*)src;
    const int chunks = u16count >> 3;   // 16B chunks
    for (int i = tid; i < chunks; i += NTHREADS) {
        unsigned int l = lds_base + (unsigned int)i * 16u;
        const void*  a = g + (size_t)i * 16;
        asm volatile("global_load_async_to_lds_b128 %0, %1, off"
                     :: "v"(l), "v"(a) : "memory");
    }
}

__device__ __forceinline__ void wait_async_all() {
    asm volatile("s_wait_asynccnt 0x0" ::: "memory");
}

__device__ __forceinline__ float sigmoidf_(float x) {
    return 1.0f / (1.0f + __expf(-x));
}

// -------- prep: f32 -> bf16 direct + transposed copies --------
__global__ void convert_bf16_kernel(const float* __restrict__ src,
                                    unsigned short* __restrict__ dst,
                                    unsigned short* __restrict__ dstT,
                                    int R, int C) {
    int idx = blockIdx.x * blockDim.x + threadIdx.x;
    if (idx >= R * C) return;
    int r = idx / C, c = idx - r * C;
    unsigned short v = f2bf(src[idx]);
    dst[idx] = v;
    dstT[(size_t)c * R + r] = v;
}

// 16 K-step WMMA chain with software-pipelined B-fragment loads.
__device__ __forceinline__ v8f gemm16_pipelined(const v16bf* afr,
                                                const unsigned short* panel,
                                                int lane, v8f c) {
    v16bf bc = lds_bfrag(panel, N_DIM, lane, 0, 0);
#pragma unroll
    for (int kk = 0; kk < 16; ++kk) {
        if (kk + 1 < 16) {
            v16bf bn = lds_bfrag(panel, N_DIM, lane, 0, (kk + 1) * 32);
            c  = wmma_bf16(afr[kk], bc, c);
            bc = bn;
        } else {
            c = wmma_bf16(afr[kk], bc, c);
        }
    }
    return c;
}

// Per-wave MLP fwd + bwd-to-d_a1 for one 16-row tile (block-cooperative weight
// staging: all waves run the same t-schedule).
__device__ __forceinline__ void mlp_d_a1(
        const unsigned short* Xin,                  // 16x512 bf16 LDS (per wave)
        const unsigned short* __restrict__ w1T,     // global, P[n][k]=w1[k][n], K=512
        const unsigned short* __restrict__ w2T,     // global, P[n][k]=w2[k][n], K=64
        const unsigned short* __restrict__ w2d,     // global, P[n][k]=w2[n][k], K=64
        const float* __restrict__ b1,
        const float* __restrict__ b2,
        const float* __restrict__ w3,
        float* a1, unsigned short* h1,
        unsigned short* da2, unsigned short* da1,
        unsigned short* panelA,                     // 16KB shared panel
        unsigned short* panelB,                     // 8KB shared panel
        int lane, int tid) {
    const int nl = lane & 15;
    const int hi = (lane >> 4) & 1;

    // ---- a1 = x @ w1 + b1 ; h1 = silu(a1) ----
    v16bf afr[16];
#pragma unroll
    for (int kk = 0; kk < 16; ++kk)
        afr[kk] = lds_afrag(Xin, N_DIM, lane, kk * 32);

    for (int t = 0; t < 4; ++t) {
        const int n0 = t * 16;
        stage_panel(panelA, w1T + (size_t)n0 * N_DIM, 16 * N_DIM, tid);
        __syncthreads();
        v8f c = {};
        c = gemm16_pipelined(afr, panelA, lane, c);
        const float bias = b1[n0 + nl];
#pragma unroll
        for (int v = 0; v < 8; ++v) {
            const int M = v + hi * 8;
            float a = c[v] + bias;
            a1[M * H_DIM + n0 + nl] = a;
            h1[M * H_DIM + n0 + nl] = f2bf(a * sigmoidf_(a));
        }
        __syncthreads();
    }

    // ---- a2 = h1 @ w2 + b2 ; d_a2 = w3 * silu'(a2) ----
    stage_panel(panelB, w2T, H_DIM * H_DIM, tid);
    __syncthreads();
    {
        v16bf h0  = lds_afrag(h1, H_DIM, lane, 0);
        v16bf h32 = lds_afrag(h1, H_DIM, lane, 32);
#pragma unroll
        for (int t = 0; t < 4; ++t) {
            const int n0 = t * 16;
            v16bf bA = lds_bfrag(panelB, H_DIM, lane, n0, 0);
            v16bf bB = lds_bfrag(panelB, H_DIM, lane, n0, 32);
            v8f c = {};
            c = wmma_bf16(h0, bA, c);
            c = wmma_bf16(h32, bB, c);
            const float bias = b2[n0 + nl];
            const float g2   = w3[n0 + nl];
#pragma unroll
            for (int v = 0; v < 8; ++v) {
                const int M = v + hi * 8;
                float a2 = c[v] + bias;
                float s  = sigmoidf_(a2);
                float ds = s * (1.0f + a2 * (1.0f - s));   // silu'
                da2[M * H_DIM + n0 + nl] = f2bf(g2 * ds);
            }
        }
    }
    __syncthreads();

    // ---- d_h1 = d_a2 @ w2^T ; d_a1 = d_h1 * silu'(a1) ----
    stage_panel(panelB, w2d, H_DIM * H_DIM, tid);
    __syncthreads();
    {
        v16bf d0  = lds_afrag(da2, H_DIM, lane, 0);
        v16bf d32 = lds_afrag(da2, H_DIM, lane, 32);
#pragma unroll
        for (int t = 0; t < 4; ++t) {
            const int n0 = t * 16;
            v16bf bA = lds_bfrag(panelB, H_DIM, lane, n0, 0);
            v16bf bB = lds_bfrag(panelB, H_DIM, lane, n0, 32);
            v8f c = {};
            c = wmma_bf16(d0, bA, c);
            c = wmma_bf16(d32, bB, c);
#pragma unroll
            for (int v = 0; v < 8; ++v) {
                const int M = v + hi * 8;
                float a  = a1[M * H_DIM + n0 + nl];
                float s  = sigmoidf_(a);
                float ds = s * (1.0f + a * (1.0f - s));
                da1[M * H_DIM + n0 + nl] = f2bf(c[v] * ds);
            }
        }
    }
    __syncthreads();
}

// Block-cooperative output half: out[:, colBase + n] = -( X @ Wpanels + dA @ w1panels )
// with async double-buffered panel staging.
__device__ __forceinline__ void output_half(
        const unsigned short* Xlds,                 // per-wave 16x512 A tile
        const unsigned short* dAlds,                // per-wave 16x64 d_a1 tile
        const unsigned short* __restrict__ Wpk,     // packed W panels, P[n][k], K=512
        const unsigned short* __restrict__ w1pk,    // packed w1 panels, P[n][k], K=64
        float* __restrict__ out, int b0, int colBase,
        unsigned short* bufA0, unsigned short* bufA1,
        unsigned short* bufB0, unsigned short* bufB1,
        int lane, int tid) {
    const int nl = lane & 15;
    const int hi = (lane >> 4) & 1;

    v16bf afr[16];
#pragma unroll
    for (int kk = 0; kk < 16; ++kk)
        afr[kk] = lds_afrag(Xlds, N_DIM, lane, kk * 32);
    v16bf ad0 = lds_afrag(dAlds, H_DIM, lane, 0);
    v16bf ad1 = lds_afrag(dAlds, H_DIM, lane, 32);

    unsigned short* bufA[2] = {bufA0, bufA1};
    unsigned short* bufB[2] = {bufB0, bufB1};

    stage_panel_async(bufA[0], Wpk, 16 * N_DIM, tid);
    stage_panel_async(bufB[0], w1pk, 16 * H_DIM, tid);
    wait_async_all();
    __syncthreads();

    for (int it = 0; it < N_DIM / 16; ++it) {
        const int n0  = it * 16;
        const int cur = it & 1;
        if (it + 1 < N_DIM / 16) {
            stage_panel_async(bufA[cur ^ 1], Wpk + (size_t)(n0 + 16) * N_DIM,
                              16 * N_DIM, tid);
            stage_panel_async(bufB[cur ^ 1], w1pk + (size_t)(n0 + 16) * H_DIM,
                              16 * H_DIM, tid);
        }
        if (it + 2 < N_DIM / 16)
            __builtin_prefetch((const char*)(Wpk + (size_t)(n0 + 32) * N_DIM)
                                   + tid * 128, 0, 1);

        v8f c = {};
        c = gemm16_pipelined(afr, bufA[cur], lane, c);
        c = wmma_bf16(ad0, lds_bfrag(bufB[cur], H_DIM, lane, 0, 0),  c);
        c = wmma_bf16(ad1, lds_bfrag(bufB[cur], H_DIM, lane, 0, 32), c);
#pragma unroll
        for (int v = 0; v < 8; ++v) {
            const int M = v + hi * 8;
            out[(size_t)(b0 + M) * (2 * N_DIM) + colBase + n0 + nl] = -c[v];
        }
        wait_async_all();
        __syncthreads();
    }
}

// Per-wave LDS (u16): a1 f32 16x64 -> 2048 ; Xp,Xf bf16 16x512 -> 2*8192 ;
// h1, da2, da1_0, da1_1 bf16 16x64 -> 4*1024
#define PER_WAVE_U16 (2048 + 8192 + 8192 + 4 * 1024)
#define PANEL_A_U16 (2 * 8192)   // double-buffered 16KB W panels
#define PANEL_B_U16 4096         // w2 staging / double-buffered 2KB w1 panels
#define TOTAL_LDS_BYTES ((WAVES * PER_WAVE_U16 + PANEL_A_U16 + PANEL_B_U16) * 2)

__global__ __launch_bounds__(NTHREADS)
void fused_score_grad(const float* __restrict__ z,
                      const unsigned short* __restrict__ Wbf,   // P[n][k]=W[n][k]
                      const unsigned short* __restrict__ WTbf,  // P[n][k]=W[k][n]
                      const unsigned short* __restrict__ w1bf0,
                      const unsigned short* __restrict__ w1Tbf0,
                      const unsigned short* __restrict__ w2bf0,
                      const unsigned short* __restrict__ w2Tbf0,
                      const float* __restrict__ b1_0, const float* __restrict__ b2_0,
                      const float* __restrict__ w3_0,
                      const unsigned short* __restrict__ w1bf1,
                      const unsigned short* __restrict__ w1Tbf1,
                      const unsigned short* __restrict__ w2bf1,
                      const unsigned short* __restrict__ w2Tbf1,
                      const float* __restrict__ b1_1, const float* __restrict__ b2_1,
                      const float* __restrict__ w3_1,
                      float* __restrict__ out) {
    extern __shared__ unsigned short smem[];
    const int tid  = threadIdx.x;
    const int lane = tid & 31;
    const int wave = tid >> 5;
    const int b0   = (blockIdx.x * WAVES + wave) * 16;

    unsigned short* base   = smem + (size_t)wave * PER_WAVE_U16;
    float*          a1     = (float*)base;
    unsigned short* Xp     = base + 2048;
    unsigned short* Xf     = Xp + 8192;
    unsigned short* h1     = Xf + 8192;
    unsigned short* da2    = h1 + 1024;
    unsigned short* da1_0  = da2 + 1024;
    unsigned short* da1_1  = da1_0 + 1024;
    unsigned short* panelA = smem + (size_t)WAVES * PER_WAVE_U16;
    unsigned short* panelB = panelA + PANEL_A_U16;

    // ---- stage z tile into LDS as bf16 (past / future halves) ----
    for (int q = lane; q < 16 * 128; q += 32) {
        const int r  = q >> 7;
        const int cq = (q & 127) * 4;
        const float4 p = *(const float4*)(z + (size_t)(b0 + r) * (2 * N_DIM) + cq);
        const float4 f = *(const float4*)(z + (size_t)(b0 + r) * (2 * N_DIM) + N_DIM + cq);
        unsigned int* dp = (unsigned int*)(Xp + r * N_DIM + cq);
        unsigned int* df = (unsigned int*)(Xf + r * N_DIM + cq);
        dp[0] = (unsigned int)f2bf(p.x) | ((unsigned int)f2bf(p.y) << 16);
        dp[1] = (unsigned int)f2bf(p.z) | ((unsigned int)f2bf(p.w) << 16);
        df[0] = (unsigned int)f2bf(f.x) | ((unsigned int)f2bf(f.y) << 16);
        df[1] = (unsigned int)f2bf(f.z) | ((unsigned int)f2bf(f.w) << 16);
    }
    __syncthreads();

    // ---- MLP input-gradients for both branches ----
    mlp_d_a1(Xp, w1Tbf0, w2Tbf0, w2bf0, b1_0, b2_0, w3_0,
             a1, h1, da2, da1_0, panelA, panelB, lane, tid);
    mlp_d_a1(Xf, w1Tbf1, w2Tbf1, w2bf1, b1_1, b2_1, w3_1,
             a1, h1, da2, da1_1, panelA, panelB, lane, tid);

    // ---- past half: out[:, :N] = -( Xf @ W + d_a1_0 @ w1_0^T ) ----
    output_half(Xf, da1_0, WTbf, w1bf0, out, b0, 0,
                panelA, panelA + 8192, panelB, panelB + 1024, lane, tid);

    // ---- future half: out[:, N:] = -( Xp @ W^T + d_a1_1 @ w1_1^T ) ----
    output_half(Xp, da1_1, Wbf, w1bf1, out, b0, N_DIM,
                panelA, panelA + 8192, panelB, panelB + 1024, lane, tid);
}

extern "C" void kernel_launch(void* const* d_in, const int* in_sizes, int n_in,
                              void* d_out, int out_size, void* d_ws, size_t ws_size,
                              hipStream_t stream) {
    const float* z     = (const float*)d_in[0];
    const float* W     = (const float*)d_in[1];
    const float* g0_w1 = (const float*)d_in[2];
    const float* g0_b1 = (const float*)d_in[3];
    const float* g0_w2 = (const float*)d_in[4];
    const float* g0_b2 = (const float*)d_in[5];
    const float* g0_w3 = (const float*)d_in[6];
    const float* g1_w1 = (const float*)d_in[8];
    const float* g1_b1 = (const float*)d_in[9];
    const float* g1_w2 = (const float*)d_in[10];
    const float* g1_b2 = (const float*)d_in[11];
    const float* g1_w3 = (const float*)d_in[12];
    float* out = (float*)d_out;

    size_t off = 0;
    auto carve = [&](size_t elems) {
        unsigned short* p = (unsigned short*)((char*)d_ws + off);
        off += ((elems * sizeof(unsigned short)) + 255) & ~(size_t)255;
        return p;
    };
    unsigned short* Wbf    = carve((size_t)N_DIM * N_DIM);
    unsigned short* WTbf   = carve((size_t)N_DIM * N_DIM);
    unsigned short* w1bf0  = carve((size_t)N_DIM * H_DIM);
    unsigned short* w1Tbf0 = carve((size_t)H_DIM * N_DIM);
    unsigned short* w2bf0  = carve((size_t)H_DIM * H_DIM);
    unsigned short* w2Tbf0 = carve((size_t)H_DIM * H_DIM);
    unsigned short* w1bf1  = carve((size_t)N_DIM * H_DIM);
    unsigned short* w1Tbf1 = carve((size_t)H_DIM * N_DIM);
    unsigned short* w2bf1  = carve((size_t)H_DIM * H_DIM);
    unsigned short* w2Tbf1 = carve((size_t)H_DIM * H_DIM);

    convert_bf16_kernel<<<(N_DIM * N_DIM + 255) / 256, 256, 0, stream>>>(
        W, Wbf, WTbf, N_DIM, N_DIM);
    convert_bf16_kernel<<<(N_DIM * H_DIM + 255) / 256, 256, 0, stream>>>(
        g0_w1, w1bf0, w1Tbf0, N_DIM, H_DIM);
    convert_bf16_kernel<<<(H_DIM * H_DIM + 255) / 256, 256, 0, stream>>>(
        g0_w2, w2bf0, w2Tbf0, H_DIM, H_DIM);
    convert_bf16_kernel<<<(N_DIM * H_DIM + 255) / 256, 256, 0, stream>>>(
        g1_w1, w1bf1, w1Tbf1, N_DIM, H_DIM);
    convert_bf16_kernel<<<(H_DIM * H_DIM + 255) / 256, 256, 0, stream>>>(
        g1_w2, w2bf1, w2Tbf1, H_DIM, H_DIM);

    const size_t shmem = TOTAL_LDS_BYTES;
    (void)hipFuncSetAttribute((const void*)fused_score_grad,
                              hipFuncAttributeMaxDynamicSharedMemorySize,
                              (int)shmem);
    const int blocks = B_DIM / (WAVES * 16);   // 512
    fused_score_grad<<<blocks, NTHREADS, shmem, stream>>>(
        z, Wbf, WTbf,
        w1bf0, w1Tbf0, w2bf0, w2Tbf0, g0_b1, g0_b2, g0_w3,
        w1bf1, w1Tbf1, w2bf1, w2Tbf1, g1_b1, g1_b2, g1_w3,
        out);
}
